// SparseAudioModel_29764123361394
// MI455X (gfx1250) — matrix operand order repaired and checked by reading, not compiled, not run on hardware
//
#include <hip/hip_runtime.h>
#include <math.h>

typedef __attribute__((ext_vector_type(16))) _Float16 v16h;
typedef __attribute__((ext_vector_type(8)))  float    v8f;

#define B_SZ      64
#define T_LEN     128
#define NATOMS    2048
#define ASIZE     2048
#define KTOP      32
#define NSAMP     32768
#define UFLAT     (NATOMS * T_LEN)   /* 262144 */

// ---------------------------------------------------------------------------
// One-time cast + transpose: x [B][1024][128] f32 -> xh [B][128][1024] f16
// (t-major so K-dim pairs are contiguous for the async B-fragment staging).
// ---------------------------------------------------------------------------
__global__ __launch_bounds__(256)
void cast_transpose_x(const float* __restrict__ x, _Float16* __restrict__ xh)
{
    const int n = B_SZ * 1024 * T_LEN;
    for (int i = blockIdx.x * 256 + threadIdx.x; i < n; i += gridDim.x * 256) {
        const int b  = i / (1024 * T_LEN);
        const int r  = i % (1024 * T_LEN);
        const int ci = r / T_LEN;
        const int t  = r % T_LEN;
        xh[((size_t)b * T_LEN + t) * 1024 + ci] = (_Float16)x[i];
    }
}

// ---------------------------------------------------------------------------
// One-time weight swizzle: W [Cout][CIN][KW] f32 -> Wf f16 in WMMA A-fragment
// order.  Fragment id = ((cot*(CIN/128)+cc)*KW + j)*4 + k4; within a fragment,
// lane l stores its 16 v16h elements contiguously (32B per lane -> the conv
// kernel loads each A fragment with one aligned 32-byte vector load).
// ---------------------------------------------------------------------------
__global__ __launch_bounds__(256)
void swizzle_w(const float* __restrict__ W, _Float16* __restrict__ Wf,
               int Cout, int CIN, int KW)
{
    const int n   = Cout * CIN * KW;
    const int nCC = CIN / 128;
    for (int i = blockIdx.x * 256 + threadIdx.x; i < n; i += gridDim.x * 256) {
        const int e    = i & 15;
        const int lane = (i >> 4) & 31;
        int frag = i >> 9;
        const int k4 = frag & 3;  frag >>= 2;
        const int j  = frag % KW; frag /= KW;
        const int cc = frag % nCC;
        const int cot = frag / nCC;
        const int lm = lane & 15, hh = lane >> 4;
        const int v  = e >> 1;
        const int kk = ((v < 4) ? (8 * hh + 2 * v)
                                : (16 + 8 * hh + 2 * (v - 4))) + (e & 1);
        const int co = cot * 16 + lm;
        const int ci = cc * 128 + k4 * 32 + kk;
        Wf[i] = (_Float16)W[((size_t)co * CIN + ci) * KW + j];
    }
}

// ---------------------------------------------------------------------------
// Conv-as-GEMM with WMMA + async global->LDS staging + pre-swizzled weights.
//   Y[b,co,t] = act(sum_{ci,j} W[co,ci,j] * X[b,ci,t+DIL*j-PAD] + bias (+Res))
// Inner loop per WMMA: one 32B A-fragment load (f16, fragment-ordered),
// four ds_load_2addr_b32 for the B fragment, one v_wmma_f32_16x16x32_f16.
// ---------------------------------------------------------------------------
template<int CIN, int KW, int DIL, int PAD, bool RES_LEAKY>
__global__ __launch_bounds__(256)
void conv_gemm(const _Float16* __restrict__ Xh, const _Float16* __restrict__ Wf,
               const float* __restrict__ bias, const float* __restrict__ Res,
               float* __restrict__ Y, _Float16* __restrict__ Yh, int Cout)
{
    constexpr int LPAD  = (PAD & 1) ? PAD + 1 : PAD;  // even leading pad (align)
    constexpr int TROWS = LPAD + T_LEN + PAD;
    constexpr int CSTR  = 130;                        // even row stride (f16)
    __shared__ _Float16 sx[TROWS * CSTR];

    const int tiles = Cout >> 4;
    const int b     = blockIdx.x / tiles;
    const int co0   = (blockIdx.x % tiles) << 4;
    const int tid   = threadIdx.x;
    const int wave  = tid >> 5;
    const int lane  = tid & 31;
    const int lm    = lane & 15;
    const int hh    = lane >> 4;
    const int t0    = wave << 4;

    // zero the pad rows once (fragments only touch ci < 128)
    if constexpr (PAD > 0) {
        for (int i = tid; i < (LPAD + PAD) * 128; i += 256) {
            const int pr  = i >> 7;
            const int ci  = i & 127;
            const int row = (pr < LPAD) ? pr : (pr + 128);
            sx[row * CSTR + ci] = (_Float16)0.f;
        }
    }

    const unsigned sxbase = (unsigned)(uintptr_t)sx;  // low 32b = LDS byte addr
    v8f acc = {};

    for (int cc = 0; cc < CIN / 128; ++cc) {
        // async stage interior rows: 128 t-rows x 64 dword granules (2 f16)
        for (int i = tid; i < 128 * 64; i += 256) {
            const int t = i >> 6, q = i & 63;
            const unsigned long long ga = (unsigned long long)(uintptr_t)
                (Xh + ((size_t)(b * T_LEN + t) * CIN + cc * 128 + 2 * q));
            const unsigned la =
                sxbase + (unsigned)(((LPAD + t) * CSTR + 2 * q) * 2);
            asm volatile("global_load_async_to_lds_b32 %0, %1, off"
                         :: "v"(la), "v"(ga) : "memory");
        }
        asm volatile("s_wait_asynccnt 0x0" ::: "memory");
        __syncthreads();

        for (int j = 0; j < KW; ++j) {
            const int row0 = LPAD - PAD + j * DIL;   // LDS row of t=0 for tap j
            for (int k0 = 0; k0 < 128; k0 += 32) {
                // A fragment: pre-swizzled f16, 32B contiguous per lane
                const size_t fidx =
                    ((((size_t)(co0 >> 4)) * (CIN / 128) + cc) * KW + j) * 4 +
                    (k0 >> 5);
                const v16h a =
                    *(const v16h*)(Wf + fidx * 512 + (size_t)lane * 16);
                // B fragment (32 k x 16 t): contiguous K-pairs in LDS
                v16h bf;
                const int roff = (row0 + t0 + lm) * CSTR;
                #pragma unroll
                for (int v = 0; v < 8; ++v) {
                    const int kk = k0 + ((v < 4) ? (8 * hh + 2 * v)
                                                 : (16 + 8 * hh + 2 * (v - 4)));
                    bf[2 * v]     = sx[roff + kk];
                    bf[2 * v + 1] = sx[roff + kk + 1];
                }
                acc = __builtin_amdgcn_wmma_f32_16x16x32_f16(
                    false, a, false, bf, (short)0, acc, false, false);
            }
        }
        __syncthreads();
    }

    // epilogue: C layout -> VGPR r holds (co0 + r + 8*hh, t0 + lm)
    #pragma unroll
    for (int r = 0; r < 8; ++r) {
        const int co = co0 + 8 * hh + r;
        const int t  = t0 + lm;
        float val = acc[r] + bias[co];
        if constexpr (RES_LEAKY) {
            val += Res[((size_t)(b * 128 + co)) * T_LEN + t];
            val = val > 0.f ? val : 0.2f * val;
        }
        Y[((size_t)b * Cout + co) * T_LEN + t] = val;
        if (Yh)
            Yh[((size_t)b * T_LEN + t) * Cout + co] = (_Float16)val;
    }
}

// ---------------------------------------------------------------------------
// Per-batch softmax statistics over flat u (max and sum of exp).
// ---------------------------------------------------------------------------
__global__ __launch_bounds__(256)
void softmax_stats(const float* __restrict__ u, float* __restrict__ stats)
{
    const int b = blockIdx.x, tid = threadIdx.x;
    __shared__ float red[256];
    const float* ub = u + (size_t)b * UFLAT;

    float m = -__builtin_inff();
    for (int i = tid; i < UFLAT; i += 256) m = fmaxf(m, ub[i]);
    red[tid] = m; __syncthreads();
    for (int s = 128; s > 0; s >>= 1) {
        if (tid < s) red[tid] = fmaxf(red[tid], red[tid + s]);
        __syncthreads();
    }
    const float M = red[0]; __syncthreads();

    float sum = 0.f;
    for (int i = tid; i < UFLAT; i += 256) sum += __expf(ub[i] - M);
    red[tid] = sum; __syncthreads();
    for (int s = 128; s > 0; s >>= 1) {
        if (tid < s) red[tid] += red[tid + s];
        __syncthreads();
    }
    if (tid == 0) { stats[b * 2] = M; stats[b * 2 + 1] = red[0]; }
}

// ---------------------------------------------------------------------------
// Top-32 per batch, mutation-free lexicographic descending scan
// (value desc, index asc == jax.lax.top_k ordering).
// ---------------------------------------------------------------------------
__global__ __launch_bounds__(256)
void topk32(const float* __restrict__ u, const float* __restrict__ stats,
            int* __restrict__ idxs, float* __restrict__ vals)
{
    const int b = blockIdx.x, tid = threadIdx.x;
    __shared__ float sv[256]; __shared__ int si[256];
    __shared__ float s_pv;    __shared__ int s_pi;
    const float* ub = u + (size_t)b * UFLAT;
    const float M = stats[b * 2], S = stats[b * 2 + 1];

    if (tid == 0) { s_pv = __builtin_inff(); s_pi = -1; }
    __syncthreads();

    for (int k = 0; k < KTOP; ++k) {
        const float pv = s_pv; const int pi = s_pi;
        float bv = -__builtin_inff(); int bi = 0x7fffffff;
        for (int i = tid; i < UFLAT; i += 256) {
            const float v = ub[i];
            const bool after = (v < pv) || (v == pv && i > pi);
            if (after && (v > bv || (v == bv && i < bi))) { bv = v; bi = i; }
        }
        sv[tid] = bv; si[tid] = bi; __syncthreads();
        for (int s = 128; s > 0; s >>= 1) {
            if (tid < s) {
                if (sv[tid + s] > sv[tid] ||
                    (sv[tid + s] == sv[tid] && si[tid + s] < si[tid])) {
                    sv[tid] = sv[tid + s]; si[tid] = si[tid + s];
                }
            }
            __syncthreads();
        }
        if (tid == 0) {
            s_pv = sv[0]; s_pi = si[0];
            idxs[b * KTOP + k] = si[0];
            vals[b * KTOP + k] = __expf(sv[0] - M) / S;
        }
        __syncthreads();
    }
}

// ---------------------------------------------------------------------------
// Sparse tv-conv: only the K selected (atom,t) positions need v = |tv(u)|.
// ---------------------------------------------------------------------------
__global__ __launch_bounds__(256)
void sel_amp(const float* __restrict__ u, const float* __restrict__ tv_w,
             const float* __restrict__ tv_b, const int* __restrict__ idxs,
             const float* __restrict__ vals, float* __restrict__ amp,
             int* __restrict__ aidx, int* __restrict__ stw)
{
    const int b = blockIdx.x / KTOP, k = blockIdx.x % KTOP;
    const int tid = threadIdx.x;
    __shared__ float red[256];
    const int idx = idxs[b * KTOP + k];
    const int a   = idx >> 7;
    const int t   = idx & 127;

    float s = 0.f;
    for (int ci = tid; ci < NATOMS; ci += 256)
        s += tv_w[(size_t)a * NATOMS + ci] * u[((size_t)b * NATOMS + ci) * T_LEN + t];
    red[tid] = s; __syncthreads();
    for (int st = 128; st > 0; st >>= 1) {
        if (tid < st) red[tid] += red[tid + st];
        __syncthreads();
    }
    if (tid == 0) {
        const float v = fabsf(red[0] + tv_b[a]);
        amp[b * KTOP + k]  = vals[b * KTOP + k] * v;
        aidx[b * KTOP + k] = a;
        stw[b * KTOP + k]  = t * (NSAMP / T_LEN);
    }
}

__global__ __launch_bounds__(256)
void atom_norms(const float* __restrict__ atoms, float* __restrict__ ninv)
{
    const int a = blockIdx.x, tid = threadIdx.x;
    __shared__ float red[256];
    float s = 0.f;
    for (int i = tid; i < ASIZE; i += 256) {
        const float v = atoms[(size_t)a * ASIZE + i];
        s += v * v;
    }
    red[tid] = s; __syncthreads();
    for (int st = 128; st > 0; st >>= 1) {
        if (tid < st) red[tid] += red[tid + st];
        __syncthreads();
    }
    if (tid == 0) ninv[a] = 1.0f / (sqrtf(red[0]) + 1e-8f);
}

__global__ __launch_bounds__(256)
void zero_out(float* __restrict__ out, int n)
{
    for (int i = blockIdx.x * 256 + threadIdx.x; i < n; i += gridDim.x * 256)
        out[i] = 0.f;
}

__global__ __launch_bounds__(256)
void scatter_add(const float* __restrict__ atoms, const float* __restrict__ ninv,
                 const float* __restrict__ amp, const int* __restrict__ aidx,
                 const int* __restrict__ stw, float* __restrict__ out)
{
    const int b = blockIdx.x / KTOP, k = blockIdx.x % KTOP;
    const int a  = aidx[b * KTOP + k];
    const int st = stw[b * KTOP + k];
    const float am = amp[b * KTOP + k] * ninv[a];
    float* ob = out + (size_t)b * NSAMP;
    for (int i = threadIdx.x; i < ASIZE; i += 256) {
        const int pos = st + i;
        if (pos < NSAMP)
            atomicAdd(&ob[pos], atoms[(size_t)a * ASIZE + i] * am);
    }
}

// ---------------------------------------------------------------------------
extern "C" void kernel_launch(void* const* d_in, const int* in_sizes, int n_in,
                              void* d_out, int out_size, void* d_ws, size_t ws_size,
                              hipStream_t stream)
{
    const float* x      = (const float*)d_in[0];
    const float* conv_w = (const float*)d_in[1];
    const float* conv_b = (const float*)d_in[2];
    const float* d1_w   = (const float*)d_in[3];
    const float* d1_b   = (const float*)d_in[4];
    const float* p1_w   = (const float*)d_in[5];
    const float* p1_b   = (const float*)d_in[6];
    const float* d3_w   = (const float*)d_in[7];
    const float* d3_b   = (const float*)d_in[8];
    const float* p3_w   = (const float*)d_in[9];
    const float* p3_b   = (const float*)d_in[10];
    const float* d9_w   = (const float*)d_in[11];
    const float* d9_b   = (const float*)d_in[12];
    const float* p9_w   = (const float*)d_in[13];
    const float* p9_b   = (const float*)d_in[14];
    const float* up_w   = (const float*)d_in[15];
    const float* up_b   = (const float*)d_in[16];
    const float* tv_w   = (const float*)d_in[17];
    const float* tv_b   = (const float*)d_in[18];
    const float* atoms  = (const float*)d_in[19];

    // workspace layout (f32 region, then 32B-aligned f16 region)
    float* bufA  = (float*)d_ws;                        // 64*128*128
    float* bufB  = bufA + (size_t)B_SZ * 128 * T_LEN;   // 64*128*128
    float* u     = bufB + (size_t)B_SZ * 128 * T_LEN;   // 64*2048*128
    float* stats = u + (size_t)B_SZ * NATOMS * T_LEN;   // 64*2
    int*   idxs  = (int*)(stats + 2 * B_SZ);
    float* vals  = (float*)(idxs + B_SZ * KTOP);
    float* amp   = vals + B_SZ * KTOP;
    int*   aidx  = (int*)(amp + B_SZ * KTOP);
    int*   stw   = aidx + B_SZ * KTOP;
    float* ninv  = (float*)(stw + B_SZ * KTOP);         // 2048
    _Float16* xh  = (_Float16*)(ninv + NATOMS);         // 64*128*1024
    _Float16* hAh = xh + (size_t)B_SZ * T_LEN * 1024;   // 64*128*128
    _Float16* hBh = hAh + (size_t)B_SZ * T_LEN * 128;   // 64*128*128
    _Float16* wfC  = hBh + (size_t)B_SZ * T_LEN * 128;  // 128*1024
    _Float16* wfD1 = wfC  + 128 * 1024;                 // 128*128*3
    _Float16* wfP1 = wfD1 + 128 * 128 * 3;              // 128*128
    _Float16* wfD3 = wfP1 + 128 * 128;
    _Float16* wfP3 = wfD3 + 128 * 128 * 3;
    _Float16* wfD9 = wfP3 + 128 * 128;
    _Float16* wfP9 = wfD9 + 128 * 128 * 3;
    _Float16* wfUp = wfP9 + 128 * 128;                  // 2048*128*7

    const dim3 blk(256);

    cast_transpose_x<<<4096, blk, 0, stream>>>(x, xh);
    swizzle_w<<< 512, blk, 0, stream>>>(conv_w, wfC,  128, 1024, 1);
    swizzle_w<<< 192, blk, 0, stream>>>(d1_w,   wfD1, 128, 128, 3);
    swizzle_w<<<  64, blk, 0, stream>>>(p1_w,   wfP1, 128, 128, 1);
    swizzle_w<<< 192, blk, 0, stream>>>(d3_w,   wfD3, 128, 128, 3);
    swizzle_w<<<  64, blk, 0, stream>>>(p3_w,   wfP3, 128, 128, 1);
    swizzle_w<<< 192, blk, 0, stream>>>(d9_w,   wfD9, 128, 128, 3);
    swizzle_w<<<  64, blk, 0, stream>>>(p9_w,   wfP9, 128, 128, 1);
    swizzle_w<<<2048, blk, 0, stream>>>(up_w,   wfUp, NATOMS, 128, 7);

    // stem + dilated blocks (Cout=128 -> 8 co-tiles per batch)
    conv_gemm<1024, 1, 1, 0, false><<<B_SZ * 8, blk, 0, stream>>>(xh,  wfC,  conv_b, nullptr, bufA, hAh, 128);
    conv_gemm< 128, 3, 1, 1, false><<<B_SZ * 8, blk, 0, stream>>>(hAh, wfD1, d1_b,   nullptr, bufB, hBh, 128);
    conv_gemm< 128, 1, 1, 0, true ><<<B_SZ * 8, blk, 0, stream>>>(hBh, wfP1, p1_b,   bufA,    bufA, hAh, 128);
    conv_gemm< 128, 3, 3, 3, false><<<B_SZ * 8, blk, 0, stream>>>(hAh, wfD3, d3_b,   nullptr, bufB, hBh, 128);
    conv_gemm< 128, 1, 1, 0, true ><<<B_SZ * 8, blk, 0, stream>>>(hBh, wfP3, p3_b,   bufA,    bufA, hAh, 128);
    conv_gemm< 128, 3, 9, 9, false><<<B_SZ * 8, blk, 0, stream>>>(hAh, wfD9, d9_b,   nullptr, bufB, hBh, 128);
    conv_gemm< 128, 1, 1, 0, true ><<<B_SZ * 8, blk, 0, stream>>>(hBh, wfP9, p9_b,   bufA,    bufA, hAh, 128);
    // up-conv k=7 to 2048 channels (no f16 copy needed)
    conv_gemm< 128, 7, 1, 3, false><<<B_SZ * 128, blk, 0, stream>>>(hAh, wfUp, up_b, nullptr, u, nullptr, NATOMS);

    softmax_stats<<<B_SZ, blk, 0, stream>>>(u, stats);
    topk32<<<B_SZ, blk, 0, stream>>>(u, stats, idxs, vals);
    sel_amp<<<B_SZ * KTOP, blk, 0, stream>>>(u, tv_w, tv_b, idxs, vals, amp, aidx, stw);
    atom_norms<<<NATOMS, blk, 0, stream>>>(atoms, ninv);

    zero_out<<<2048, blk, 0, stream>>>((float*)d_out, B_SZ * NSAMP);
    scatter_add<<<B_SZ * KTOP, blk, 0, stream>>>(atoms, ninv, amp, aidx, stw, (float*)d_out);
}
